// CifMiddleware_73220602462362
// MI455X (gfx1250) — matrix operand order repaired and checked by reading, not compiled, not run on hardware
//
#include <hip/hip_runtime.h>
#include <math.h>

// ---------------------------------------------------------------------------
// CIF middleware for MI455X (gfx1250, wave32, WMMA).
//
// Pipeline:
//  K1 wt_conv    : dense_w (1024x1024 f32) -> Wt[n][k] f16 (transposed, 2MB)
//  K2 cif_gemm   : fused  w_raw[b,t] = sigmoid( sum_n relu(x_row . W[:,n] + b_n)*wv_n + wb ) * not_pad
//                  WMMA f32_16x16x32_f16; A staged in 64KB LDS in fragment
//                  order; each wave computes BOTH 16-row tiles per B fragment
//                  (2 WMMAs per Wt load -> halves L2 traffic on Wt).
//  K3 batch_stats: per-batch sum (quantity_out), scale = tlen/sum, pad_start
//  K4 cif_scan   : serial scalar integrate-and-fire per batch -> event list
//  K5 zero_mask  : zero non-emitted output rows, write padding mask (as float)
//  K6 cif_emit   : one block per fire event: weighted sum of contiguous x rows
// ---------------------------------------------------------------------------

typedef _Float16 h16 __attribute__((ext_vector_type(16)));
typedef _Float16 h8  __attribute__((ext_vector_type(8)));
typedef float    f8  __attribute__((ext_vector_type(8)));

constexpr int Bb = 32, Tt = 2000, Dd = 1024, Uu = 1024;
constexpr int MROWS = Bb * Tt;     // 64000, divisible by 32
constexpr int MAXEV = 256;         // fires per batch <= tlen < 250

// ---------------- K1: transpose + f32->f16 convert of dense_w ----------------
__global__ __launch_bounds__(256) void wt_conv(const float* __restrict__ W,
                                               _Float16* __restrict__ Wt) {
  __shared__ float tile[32][33];
  const int n0 = blockIdx.x * 32, k0 = blockIdx.y * 32;
  const int tx = threadIdx.x, ty = threadIdx.y;   // (32,8)
  for (int j = 0; j < 32; j += 8)
    tile[ty + j][tx] = W[(size_t)(k0 + ty + j) * Uu + n0 + tx];
  __syncthreads();
  for (int j = 0; j < 32; j += 8)
    Wt[(size_t)(n0 + ty + j) * Dd + k0 + tx] = (_Float16)tile[tx][ty + j];
}

// ---------------- K2: fused GEMM + relu + weighted reduce + sigmoid ----------
// Block: 128 threads = 4 waves, 32 rows of x.  LDS = 64KB A-tile in WMMA
// fragment order:  half index ((G*32+kk)*32 + lane)*16 + c*8 + j
//   lane = g*16+Lm holds A[M = G*16+Lm][K = kk*32 + g*16 + c*8 + j]  (contiguous k, same as B)
// Each wave owns 16 N-tiles (256 cols) and computes both 16-row G-groups,
// so every B fragment feeds two v_wmma ops.
__global__ __launch_bounds__(128) void cif_gemm(
    const float* __restrict__ x, const unsigned char* __restrict__ mask,
    const _Float16* __restrict__ Wt, const float* __restrict__ bias,
    const float* __restrict__ wv, const float* __restrict__ wbp,
    float* __restrict__ w_raw) {
  __shared__ __align__(32) _Float16 Atile[32 * 1024];   // 65536 bytes
  const int tid = threadIdx.x;
  const int row0 = blockIdx.x * 32;

  // ---- stage A (32 rows x 1024 K) as f16 fragments ----
  for (int it = 0; it < 32; ++it) {
    int ci = it * 128 + tid;            // 4096 chunks of 8 halves
    int m = ci >> 7, k8 = ci & 127;     // row-in-block, k/8
    const float* src = x + (size_t)(row0 + m) * Dd + k8 * 8;
    float4 p0 = *(const float4*)src;
    float4 p1 = *(const float4*)(src + 4);
    h8 hv;
    hv[0] = (_Float16)p0.x; hv[1] = (_Float16)p0.y;
    hv[2] = (_Float16)p0.z; hv[3] = (_Float16)p0.w;
    hv[4] = (_Float16)p1.x; hv[5] = (_Float16)p1.y;
    hv[6] = (_Float16)p1.z; hv[7] = (_Float16)p1.w;
    int G = m >> 4, Lm = m & 15;
    int kk = k8 >> 2, c = k8 & 1, g = (k8 >> 1) & 1;
    int dst = (((G * 32 + kk) * 32 + (g * 16 + Lm)) * 2 + c) * 8;
    *(h8*)&Atile[dst] = hv;
  }
  __syncthreads();

  const int w = tid >> 5, lane = tid & 31;
  const int Lm = lane & 15, g = lane >> 4;
  float rs0[8] = {0.f, 0.f, 0.f, 0.f, 0.f, 0.f, 0.f, 0.f};  // rows  0..15
  float rs1[8] = {0.f, 0.f, 0.f, 0.f, 0.f, 0.f, 0.f, 0.f};  // rows 16..31

  const _Float16* abase0 = &Atile[(size_t)lane * 16];                // G = 0
  const _Float16* abase1 = &Atile[32 * 512 + (size_t)lane * 16];     // G = 1

  for (int nt = 0; nt < 16; ++nt) {
    const int n = w * 256 + nt * 16 + Lm;    // wave w owns cols [w*256, w*256+256)
    const float bn = bias[n];
    const float wvn = wv[n];
    // B fragment: lane holds Wt[n][kk*32 + g*16 + (0..15)]  (contiguous 32B)
    const _Float16* bbase = Wt + (size_t)n * Dd + g * 16;
    f8 acc0 = {};
    f8 acc1 = {};
#pragma unroll 4
    for (int kk = 0; kk < 32; ++kk) {
      h16 b  = *(const h16*)(bbase + kk * 32);
      h16 a0 = *(const h16*)(abase0 + (size_t)kk * 512);
      h16 a1 = *(const h16*)(abase1 + (size_t)kk * 512);
      acc0 = __builtin_amdgcn_wmma_f32_16x16x32_f16(
          false, a0, false, b, (short)0, acc0, false, false);
      acc1 = __builtin_amdgcn_wmma_f32_16x16x32_f16(
          false, a1, false, b, (short)0, acc1, false, false);
    }
#pragma unroll
    for (int v = 0; v < 8; ++v) {
      rs0[v] += fmaxf(acc0[v] + bn, 0.f) * wvn;   // relu + weight column
      rs1[v] += fmaxf(acc1[v] + bn, 0.f) * wvn;
    }
  }

  // reduce across the 16 lanes sharing each row (C/D layout: col = Lm)
#pragma unroll
  for (int v = 0; v < 8; ++v)
    for (int off = 1; off < 16; off <<= 1) {
      rs0[v] += __shfl_xor(rs0[v], off, 32);
      rs1[v] += __shfl_xor(rs1[v], off, 32);
    }

  __syncthreads();                       // done reading Atile; reuse as scratch
  float* pf = reinterpret_cast<float*>(Atile);   // pf[w][32] partials
  if (Lm == 0) {                         // lanes 0 (g=0) and 16 (g=1)
#pragma unroll
    for (int v = 0; v < 8; ++v) {
      pf[w * 32 + g * 8 + v]      = rs0[v];     // row  g*8+v
      pf[w * 32 + 16 + g * 8 + v] = rs1[v];     // row  16+g*8+v
    }
  }
  __syncthreads();
  if (tid < 32) {                        // combine 4 wave partials per row
    float val = pf[tid] + pf[32 + tid] + pf[64 + tid] + pf[96 + tid] + wbp[0];
    const int row = row0 + tid;
    const float np = mask[row] ? 0.f : 1.f;
    w_raw[row] = np / (1.f + expf(-val));
  }
}

// ---------------- K3: per-batch sums -----------------------------------------
__global__ __launch_bounds__(256) void batch_stats(
    const float* __restrict__ w_raw, const unsigned char* __restrict__ mask,
    const int* __restrict__ tlen, float* __restrict__ quantity_out,
    float* __restrict__ scale, float* __restrict__ pad_start) {
  __shared__ float s1[256], s2[256];
  const int b = blockIdx.x, tid = threadIdx.x;
  float a = 0.f, c = 0.f;
  for (int t = tid; t < Tt; t += 256) {
    a += w_raw[b * Tt + t];
    c += mask[b * Tt + t] ? 0.f : 1.f;
  }
  s1[tid] = a; s2[tid] = c;
  __syncthreads();
  for (int s = 128; s > 0; s >>= 1) {
    if (tid < s) { s1[tid] += s1[tid + s]; s2[tid] += s2[tid + s]; }
    __syncthreads();
  }
  if (tid == 0) {
    const float q = s1[0];
    quantity_out[b] = q;                       // org_w.sum(-1)
    scale[b] = (float)tlen[b] / q;
    pad_start[b] = s2[0];                      // not_pad.sum(-1)
  }
}

// ---------------- K4: serial scalar integrate-and-fire scan ------------------
__global__ void cif_scan(const float* __restrict__ w_raw,
                         const float* __restrict__ scale,
                         const float* __restrict__ pad_start,
                         int* __restrict__ ev_t, float* __restrict__ ev_rem,
                         int* __restrict__ ev_slot, int* __restrict__ nf,
                         int* __restrict__ n_emit) {
  const int b = threadIdx.x;
  if (b >= Bb) return;
  const float sc = scale[b], ps = pad_start[b];
  float acc = 0.f;
  int e = 0, ne = 0;
  for (int t = 0; t < Tt; ++t) {
    const float wt = w_raw[b * Tt + t] * sc;
    if (acc + wt >= 1.0f) {                    // fire
      const float rem = 1.f - acc;
      if (e < MAXEV) {
        ev_t[b * MAXEV + e] = t;
        ev_rem[b * MAXEV + e] = rem;
        const bool emit = !((float)t > ps);    // reference: zero f_s if t > pad_start
        ev_slot[b * MAXEV + e] = emit ? ne : -1;
        if (emit) ne++;
      }
      acc = wt - rem;
      e++;
    } else {
      acc += wt;
    }
  }
  nf[b] = e < MAXEV ? e : MAXEV;
  n_emit[b] = ne;
}

// ---------------- K5: zero non-emitted rows + padding mask -------------------
__global__ __launch_bounds__(256) void zero_mask(const int* __restrict__ n_emit,
                                                 float* __restrict__ cif_out,
                                                 float* __restrict__ mask_out) {
  const int t = blockIdx.x, b = blockIdx.y, tid = threadIdx.x;
  const int ne = n_emit[b];
  if (t < ne) {
    if (tid == 0) mask_out[b * Tt + t] = 1.0f;
    return;                                    // row fully written by cif_emit
  }
  const float4 z = {0.f, 0.f, 0.f, 0.f};
  float4* dst = (float4*)(cif_out + ((size_t)b * Tt + t) * Dd);
  dst[tid] = z;                                // 256 x 16B = 1024 floats
  if (tid == 0) mask_out[b * Tt + t] = 0.0f;
}

// ---------------- K6: one block per emitted fire event -----------------------
__device__ __forceinline__ void fma4(float4& s, float c, const float4 v) {
  s.x += c * v.x; s.y += c * v.y; s.z += c * v.z; s.w += c * v.w;
}

__global__ __launch_bounds__(256) void cif_emit(
    const float* __restrict__ x, const float* __restrict__ w_raw,
    const float* __restrict__ scale, const int* __restrict__ ev_t,
    const float* __restrict__ ev_rem, const int* __restrict__ ev_slot,
    const int* __restrict__ nf, float* __restrict__ cif_out) {
  const int e = blockIdx.x, b = blockIdx.y, tid = threadIdx.x;
  if (e >= nf[b]) return;
  const int slot = ev_slot[b * MAXEV + e];
  if (slot < 0) return;
  const int te = ev_t[b * MAXEV + e];
  const float rem = ev_rem[b * MAXEV + e];
  const float sc = scale[b];
  const float* xb = x + (size_t)b * Tt * Dd;
  float4 s = {0.f, 0.f, 0.f, 0.f};
  int tstart = 0;
  if (e > 0) {                                 // leftover of previous fire
    const int tp = ev_t[b * MAXEV + e - 1];
    const float lf = w_raw[b * Tt + tp] * sc - ev_rem[b * MAXEV + e - 1];
    fma4(s, lf, ((const float4*)(xb + (size_t)tp * Dd))[tid]);
    tstart = tp + 1;
  }
  for (int tau = tstart; tau < te; ++tau) {    // interior accumulation
    const float cw = w_raw[b * Tt + tau] * sc;
    fma4(s, cw, ((const float4*)(xb + (size_t)tau * Dd))[tid]);
  }
  fma4(s, rem, ((const float4*)(xb + (size_t)te * Dd))[tid]);  // firing step
  ((float4*)(cif_out + ((size_t)b * Tt + slot) * Dd))[tid] = s;
}

// ---------------- launcher ---------------------------------------------------
extern "C" void kernel_launch(void* const* d_in, const int* in_sizes, int n_in,
                              void* d_out, int out_size, void* d_ws,
                              size_t ws_size, hipStream_t stream) {
  const float* x          = (const float*)d_in[0];
  const unsigned char* mk = (const unsigned char*)d_in[1];   // bool mask
  const int* tlen         = (const int*)d_in[2];
  const float* dense_w    = (const float*)d_in[3];
  const float* dense_b    = (const float*)d_in[4];
  const float* weight_w   = (const float*)d_in[5];           // (U,1)
  const float* weight_b   = (const float*)d_in[6];

  float* out          = (float*)d_out;
  float* cif_out      = out;                                 // B*T*D
  float* quantity_out = out + (size_t)Bb * Tt * Dd;          // B
  float* mask_out     = quantity_out + Bb;                   // B*T (as float)

  char* ws = (char*)d_ws;
  _Float16* Wt  = (_Float16*)ws;                             // 2 MB
  float* w_raw  = (float*)(ws + 2097152);                    // 256000 B
  float* scale  = (float*)(ws + 2353152);                    // 128 B
  float* pstart = (float*)(ws + 2353280);                    // 128 B
  int*   nf     = (int*)(ws + 2353408);                      // 128 B
  int*   nemit  = (int*)(ws + 2353536);                      // 128 B
  int*   ev_t   = (int*)(ws + 2353664);                      // 32 KB
  float* ev_rem = (float*)(ws + 2353664 + 32768);            // 32 KB
  int*   ev_slot= (int*)(ws + 2353664 + 65536);              // 32 KB

  wt_conv<<<dim3(Uu / 32, Dd / 32), dim3(32, 8), 0, stream>>>(dense_w, Wt);
  cif_gemm<<<MROWS / 32, 128, 0, stream>>>(x, mk, Wt, dense_b, weight_w,
                                           weight_b, w_raw);
  batch_stats<<<Bb, 256, 0, stream>>>(w_raw, mk, tlen, quantity_out, scale,
                                      pstart);
  cif_scan<<<1, 32, 0, stream>>>(w_raw, scale, pstart, ev_t, ev_rem, ev_slot,
                                 nf, nemit);
  zero_mask<<<dim3(Tt, Bb), 256, 0, stream>>>(nemit, cif_out, mask_out);
  cif_emit<<<dim3(MAXEV, Bb), 256, 0, stream>>>(x, w_raw, scale, ev_t, ev_rem,
                                                ev_slot, nf, cif_out);
}
